// DeformableTransformer_4818953306915
// MI455X (gfx1250) — matrix operand order repaired and checked by reading, not compiled, not run on hardware
//
#include <hip/hip_runtime.h>
#include <hip/hip_bf16.h>
#include <math.h>

// ---------------- model dims ----------------
#define Dm    256
#define NHm   8
#define NLm   4
#define NPm   4
#define DFFm  1024
#define NENCm 6
#define NDECm 8
#define NQm   300
#define Nb    2
#define Ltok  5440            // 64*64 + 32*32 + 16*16 + 8*8
#define RE    (Nb * Ltok)     // 10880 rows (multiple of 16)
#define RQ    (Nb * NQm)      // 600 rows

__device__ __constant__ int c_HL[4] = {64, 32, 16, 8};
__device__ __constant__ int c_ST[4] = {0, 4096, 5120, 5376};

typedef __bf16 bf16_t;
typedef __bf16  v16bf __attribute__((ext_vector_type(16)));
typedef __bf16  v8bf  __attribute__((ext_vector_type(8)));
typedef float   v8f   __attribute__((ext_vector_type(8)));

static inline int cdiv(int a, int b) { return (a + b - 1) / b; }

// ---------------- activation fp32 -> bf16 (same layout) ----------------
__global__ void k_acvt(const float* __restrict__ A, bf16_t* __restrict__ Ab, int n) {
  int i = blockIdx.x * blockDim.x + threadIdx.x;
  if (i < n) Ab[i] = (bf16_t)A[i];
}

// ---------------- weight fp32[K][N] -> bf16 Wt[N][K] ----------------
__global__ void k_wconv(const float* __restrict__ W, bf16_t* __restrict__ Wt, int N, int K) {
  int i = blockIdx.x * blockDim.x + threadIdx.x;
  if (i >= N * K) return;
  int n = i / K, k = i % K;
  Wt[(size_t)n * K + k] = (bf16_t)W[(size_t)k * N + n];
}

// ---------------- WMMA GEMM: C = act(A[M][K] @ W + bias) ----------------
// A in bf16 row-major, W given as Wt[N][K] bf16.
// block = 256 threads = 8 waves stacked along M; each wave computes 16 rows x 64 cols
// (4 accumulators; A fragment reused for 4 WMMAs per K-step).
__global__ void k_gemm_wmma(const bf16_t* __restrict__ A, const bf16_t* __restrict__ Wt,
                            const float* __restrict__ bias, float* __restrict__ C,
                            int M, int N, int K, int relu) {
  int lane = threadIdx.x & 31;
  int wave = threadIdx.x >> 5;
  int row0 = (blockIdx.y * 8 + wave) * 16;
  if (row0 >= M) return;                       // wave-uniform exit
  int col0 = blockIdx.x * 64;
  int g    = lane >> 4;
  int mrow = row0 + (lane & 15);
  if (mrow >= M) mrow = M - 1;                 // clamp loads; stores guarded below
  int nidx = lane & 15;

  const bf16_t* arow = A + (size_t)mrow * K;
  const bf16_t* brow0 = Wt + (size_t)(col0 + 0  + nidx) * K;
  const bf16_t* brow1 = Wt + (size_t)(col0 + 16 + nidx) * K;
  const bf16_t* brow2 = Wt + (size_t)(col0 + 32 + nidx) * K;
  const bf16_t* brow3 = Wt + (size_t)(col0 + 48 + nidx) * K;

  v8f acc0, acc1, acc2, acc3;
#pragma unroll
  for (int r = 0; r < 8; ++r) {
    acc0[r] = bias ? bias[col0 + 0  + nidx] : 0.0f;
    acc1[r] = bias ? bias[col0 + 16 + nidx] : 0.0f;
    acc2[r] = bias ? bias[col0 + 32 + nidx] : 0.0f;
    acc3[r] = bias ? bias[col0 + 48 + nidx] : 0.0f;
  }

  for (int kk = 0; kk < K; kk += 32) {
    // A fragment: element e -> K = g*16 + e  (contiguous in A row)
    v16bf a = *(const v16bf*)(arow + kk + g * 16);
    // B fragments: element e -> K = g*16 + e  (contiguous in Wt row)
    v16bf b0 = *(const v16bf*)(brow0 + kk + g * 16);
    v16bf b1 = *(const v16bf*)(brow1 + kk + g * 16);
    v16bf b2 = *(const v16bf*)(brow2 + kk + g * 16);
    v16bf b3 = *(const v16bf*)(brow3 + kk + g * 16);
    if (kk + 32 < K) __builtin_prefetch((const void*)(arow + kk + 32), 0, 0);
    acc0 = __builtin_amdgcn_wmma_f32_16x16x32_bf16(false, a, false, b0, (short)0, acc0, false, false);
    acc1 = __builtin_amdgcn_wmma_f32_16x16x32_bf16(false, a, false, b1, (short)0, acc1, false, false);
    acc2 = __builtin_amdgcn_wmma_f32_16x16x32_bf16(false, a, false, b2, (short)0, acc2, false, false);
    acc3 = __builtin_amdgcn_wmma_f32_16x16x32_bf16(false, a, false, b3, (short)0, acc3, false, false);
  }
#pragma unroll
  for (int r = 0; r < 8; ++r) {
    int row = row0 + g * 8 + r;
    if (row < M) {
      float v0 = acc0[r], v1 = acc1[r], v2 = acc2[r], v3 = acc3[r];
      if (relu) {
        v0 = fmaxf(v0, 0.0f); v1 = fmaxf(v1, 0.0f);
        v2 = fmaxf(v2, 0.0f); v3 = fmaxf(v3, 0.0f);
      }
      size_t rb = (size_t)row * N + col0 + nidx;
      C[rb + 0] = v0; C[rb + 16] = v1; C[rb + 32] = v2; C[rb + 48] = v3;
    }
  }
}

// ---------------- flatten: x(N,D,H,W) -> xf(N,L,D); pf = pos + level_embed ----------
__global__ void k_flatten(const float* x0, const float* x1, const float* x2, const float* x3,
                          const float* p0, const float* p1, const float* p2, const float* p3,
                          const unsigned char* m0, const unsigned char* m1,
                          const unsigned char* m2, const unsigned char* m3,
                          const float* lvl, float* xf, float* pf, float* mf) {
  int i = blockIdx.x * blockDim.x + threadIdx.x;
  if (i >= RE * Dm) return;
  int d   = i % Dm;
  int row = i / Dm;
  int b   = row / Ltok;
  int tok = row % Ltok;
  int l = (tok < 4096) ? 0 : (tok < 5120) ? 1 : (tok < 5376) ? 2 : 3;
  int Hl = c_HL[l], Wl = Hl;
  int li = tok - c_ST[l];
  int hy = li / Wl, wx = li % Wl;
  const float* xs = (l == 0) ? x0 : (l == 1) ? x1 : (l == 2) ? x2 : x3;
  const float* ps = (l == 0) ? p0 : (l == 1) ? p1 : (l == 2) ? p2 : p3;
  size_t src = (((size_t)b * Dm + d) * Hl + hy) * Wl + wx;
  xf[i] = xs[src];
  pf[i] = ps[src] + lvl[l * Dm + d];
  if (d == 0) {
    const unsigned char* ms = (l == 0) ? m0 : (l == 1) ? m1 : (l == 2) ? m2 : m3;
    mf[row] = ms[((size_t)b * Hl + hy) * Wl + wx] ? 1.0f : 0.0f;
  }
}

// ---------------- valid ratios vr(N,NL,2) = [vw, vh] ----------------
__global__ void k_vr(const unsigned char* m0, const unsigned char* m1,
                     const unsigned char* m2, const unsigned char* m3, float* vr) {
  int t = threadIdx.x;
  if (t >= Nb * NLm) return;
  int b = t / NLm, l = t % NLm;
  const unsigned char* m = (l == 0) ? m0 : (l == 1) ? m1 : (l == 2) ? m2 : m3;
  int H = c_HL[l], W = H;
  int ch = 0, cw = 0;
  for (int h = 0; h < H; ++h) ch += m[((size_t)b * H + h) * W] ? 0 : 1;
  for (int w = 0; w < W; ++w) cw += m[(size_t)b * H * W + w] ? 0 : 1;
  vr[(b * NLm + l) * 2 + 0] = (float)cw / (float)W;
  vr[(b * NLm + l) * 2 + 1] = (float)ch / (float)H;
}

// ---------------- encoder base reference points (N,L,2) ----------------
__global__ void k_encref(const float* vr, float* refb) {
  int i = blockIdx.x * blockDim.x + threadIdx.x;
  if (i >= RE) return;
  int b = i / Ltok, tok = i % Ltok;
  int l = (tok < 4096) ? 0 : (tok < 5120) ? 1 : (tok < 5376) ? 2 : 3;
  int Hl = c_HL[l], Wl = Hl;
  int li = tok - c_ST[l];
  int iy = li / Wl, ix = li % Wl;
  float vw = vr[(b * NLm + l) * 2 + 0];
  float vh = vr[(b * NLm + l) * 2 + 1];
  refb[(size_t)i * 2 + 0] = (ix + 0.5f) / (vw * Wl);
  refb[(size_t)i * 2 + 1] = (iy + 0.5f) / (vh * Hl);
}

// ---------------- misc elementwise ----------------
__global__ void k_add(const float* a, const float* b, float* c, int n) {
  int i = blockIdx.x * blockDim.x + threadIdx.x;
  if (i < n) c[i] = a[i] + b[i];
}
__global__ void k_copy(const float* s, float* d, int n) {
  int i = blockIdx.x * blockDim.x + threadIdx.x;
  if (i < n) d[i] = s[i];
}
__global__ void k_maskval(float* val, const float* mf, int rows) {
  int i = blockIdx.x * blockDim.x + threadIdx.x;
  if (i >= rows * Dm) return;
  if (mf[i / Dm] != 0.0f) val[i] = 0.0f;
}
__global__ void k_split(const float* qe, float* qpos, float* tgt) {
  int i = blockIdx.x * blockDim.x + threadIdx.x;
  if (i >= RQ * Dm) return;
  int row = i / Dm, d = i % Dm;
  qpos[i] = qe[(size_t)row * (2 * Dm) + d];
  tgt[i]  = qe[(size_t)row * (2 * Dm) + Dm + d];
}

// ---------------- fused residual + LayerNorm over 256 ----------------
__global__ void k_addln(const float* __restrict__ a, const float* __restrict__ b,
                        const float* __restrict__ gam, const float* __restrict__ bet,
                        float* __restrict__ y) {
  __shared__ float red[Dm];
  int row = blockIdx.x, t = threadIdx.x;
  size_t base = (size_t)row * Dm;
  float v = a[base + t] + b[base + t];
  red[t] = v; __syncthreads();
  for (int s = Dm / 2; s > 0; s >>= 1) { if (t < s) red[t] += red[t + s]; __syncthreads(); }
  float mean = red[0] * (1.0f / Dm);
  __syncthreads();
  float d = v - mean;
  red[t] = d * d; __syncthreads();
  for (int s = Dm / 2; s > 0; s >>= 1) { if (t < s) red[t] += red[t + s]; __syncthreads(); }
  float var = red[0] * (1.0f / Dm);
  y[base + t] = d * rsqrtf(var + 1e-5f) * gam[t] + bet[t];
}

// ---------------- MSDA sampling: grid = N*Lq blocks, 256 thr (wave h, lane dh) -----
__global__ void k_msda(const float* __restrict__ val, const float* __restrict__ off,
                       const float* __restrict__ awl, const float* __restrict__ refb,
                       const float* __restrict__ vr, float* __restrict__ out, int Lq) {
  int blk = blockIdx.x;
  int b = blk / Lq;
  int h = threadIdx.x >> 5, dh = threadIdx.x & 31;
  const float* lg = awl + (size_t)blk * (NHm * 16) + h * 16;
  float w[16];
  float mx = -1e30f;
#pragma unroll
  for (int i = 0; i < 16; ++i) { w[i] = lg[i]; mx = fmaxf(mx, w[i]); }
  float s = 0.0f;
#pragma unroll
  for (int i = 0; i < 16; ++i) { w[i] = expf(w[i] - mx); s += w[i]; }
  float inv = 1.0f / s;
  float rx = refb[(size_t)blk * 2 + 0], ry = refb[(size_t)blk * 2 + 1];
  const float* offp = off + (size_t)blk * Dm + h * (NLm * NPm * 2);
  float acc = 0.0f;
#pragma unroll
  for (int l = 0; l < NLm; ++l) {
    int Hl = c_HL[l], Wl = Hl, st = c_ST[l];
    float vx = vr[(b * NLm + l) * 2 + 0], vy = vr[(b * NLm + l) * 2 + 1];
    float cxb = rx * vx * Wl - 0.5f;
    float cyb = ry * vy * Hl - 0.5f;
#pragma unroll
    for (int p = 0; p < NPm; ++p) {
      float lx = cxb + offp[(l * NPm + p) * 2 + 0];
      float ly = cyb + offp[(l * NPm + p) * 2 + 1];
      float x0f = floorf(lx), y0f = floorf(ly);
      float fx = lx - x0f, fy = ly - y0f;
      int ix = (int)x0f, iy = (int)y0f;
      float samp = 0.0f;
#pragma unroll
      for (int cy = 0; cy < 2; ++cy) {
#pragma unroll
        for (int cx = 0; cx < 2; ++cx) {
          int yy = iy + cy, xx = ix + cx;
          bool ok = (yy >= 0) && (yy < Hl) && (xx >= 0) && (xx < Wl);
          int yc = min(max(yy, 0), Hl - 1), xc = min(max(xx, 0), Wl - 1);
          float v = val[(((size_t)b * Ltok + st + yc * Wl + xc) * Dm) + h * 32 + dh];
          float wgt = (cy ? fy : (1.0f - fy)) * (cx ? fx : (1.0f - fx));
          samp += ok ? v * wgt : 0.0f;
        }
      }
      acc += (w[l * NPm + p] * inv) * samp;
    }
  }
  out[(size_t)blk * Dm + h * 32 + dh] = acc;
}

// ---------------- MHA over NQ keys: one wave per (b,h,q), online softmax ----------
__global__ void k_mha(const float* __restrict__ Q, const float* __restrict__ Kb,
                      const float* __restrict__ Vb, float* __restrict__ O) {
  int blk = blockIdx.x;
  int qi = blk % NQm;
  int h  = (blk / NQm) % NHm;
  int b  = blk / (NQm * NHm);
  int lane = threadIdx.x;
  const float scale = 0.17677669529663687f;  // 1/sqrt(32)
  float qv = Q[((size_t)(b * NQm + qi)) * Dm + h * 32 + lane];
  float m = -1e30f, lsum = 0.0f, oacc = 0.0f;
  for (int k = 0; k < NQm; ++k) {
    size_t krow = ((size_t)(b * NQm + k)) * Dm + h * 32;
    float pp = qv * Kb[krow + lane];
#pragma unroll
    for (int s = 16; s > 0; s >>= 1) pp += __shfl_xor(pp, s, 32);
    float sc = pp * scale;
    float mn = fmaxf(m, sc);
    float c = expf(m - mn), e = expf(sc - mn);
    lsum = lsum * c + e;
    oacc = oacc * c + e * Vb[krow + lane];
    m = mn;
  }
  O[((size_t)(b * NQm + qi)) * Dm + h * 32 + lane] = oacc / lsum;
}

// ---------------- ref points for queries: sigmoid(qpos @ ref_w + ref_b) ----------
__global__ void k_refq(const float* qpos, const float* rw, const float* rb, float* refq) {
  int i = blockIdx.x * blockDim.x + threadIdx.x;
  if (i >= RQ * 2) return;
  int row = i >> 1, c = i & 1;
  float s = rb[c];
  for (int k = 0; k < Dm; ++k) s += qpos[(size_t)row * Dm + k] * rw[k * 2 + c];
  refq[i] = 1.0f / (1.0f + expf(-s));
}

// ---------------- write init_ref + irefs into d_out ----------------
__global__ void k_refs_out(const float* refq, float* out) {
  int i = blockIdx.x * blockDim.x + threadIdx.x;
  if (i >= RQ * 2) return;
  const int HS = NDECm * RQ * Dm;  // 1228800
  out[HS + i] = refq[i];
  for (int l = 0; l < NDECm; ++l) out[HS + RQ * 2 + l * RQ * 2 + i] = refq[i];
}

// =====================================================================
extern "C" void kernel_launch(void* const* d_in, const int* in_sizes, int n_in,
                              void* d_out, int out_size, void* d_ws, size_t ws_size,
                              hipStream_t stream) {
  (void)in_sizes; (void)n_in; (void)out_size; (void)ws_size;
  auto F = [&](int i) { return (const float*)d_in[i]; };
  auto M = [&](int i) { return (const unsigned char*)d_in[i]; };
  // input order (setup_inputs dict order): x0,pos0,mask0, x1,pos1,mask1, ...,
  // query_embed, then params (insertion order, recursively flattened)
  const float* xin[4]  = {F(0), F(3), F(6), F(9)};
  const float* pin[4]  = {F(1), F(4), F(7), F(10)};
  const unsigned char* min_[4] = {M(2), M(5), M(8), M(11)};
  const float* qembed  = F(12);
  const float* lvl_emb = F(13);
  const float* ref_w   = F(14);
  const float* ref_b   = F(15);
  const int ENC0 = 16;            // 16 leaves per encoder layer
  const int DEC0 = 16 + 6 * 16;   // 26 leaves per decoder layer

  // ---- workspace carve-up ----
  char* base = (char*)d_ws; size_t woff = 0;
  auto alloc = [&](size_t bytes) -> void* {
    void* p = base + woff; woff = (woff + bytes + 255) & ~(size_t)255; return p;
  };
  float* xf   = (float*)alloc((size_t)RE * Dm * 4);
  float* pf   = (float*)alloc((size_t)RE * Dm * 4);
  float* mf   = (float*)alloc((size_t)RE * 4);
  float* vr   = (float*)alloc(Nb * NLm * 2 * 4);
  float* refe = (float*)alloc((size_t)RE * 2 * 4);
  float* qin  = (float*)alloc((size_t)RE * Dm * 4);
  float* valb = (float*)alloc((size_t)RE * Dm * 4);
  float* offb = (float*)alloc((size_t)RE * Dm * 4);
  float* awb  = (float*)alloc((size_t)RE * (NHm * 16) * 4);
  float* samp = (float*)alloc((size_t)RE * Dm * 4);
  float* proj = (float*)alloc((size_t)RE * Dm * 4);
  float* ffh  = (float*)alloc((size_t)RE * DFFm * 4);
  float* yb   = (float*)alloc((size_t)RE * Dm * 4);
  bf16_t* wtmp = (bf16_t*)alloc((size_t)DFFm * Dm * sizeof(bf16_t));
  bf16_t* atmp = (bf16_t*)alloc((size_t)RE * DFFm * sizeof(bf16_t));
  float* qpos = (float*)alloc((size_t)RQ * Dm * 4);
  float* xdec = (float*)alloc((size_t)RQ * Dm * 4);
  float* refq = (float*)alloc((size_t)RQ * 2 * 4);
  float* qb   = (float*)alloc((size_t)RQ * Dm * 4);
  float* kb   = (float*)alloc((size_t)RQ * Dm * 4);
  float* vb   = (float*)alloc((size_t)RQ * Dm * 4);
  float* attb = (float*)alloc((size_t)RQ * Dm * 4);
  float* x2q  = (float*)alloc((size_t)RQ * Dm * 4);
  float* x3q  = (float*)alloc((size_t)RQ * Dm * 4);
  float* outp = (float*)d_out;

  auto gemm = [&](const float* A, const float* W, const float* bias, float* C,
                  int Mr, int Nc, int Kd, int relu) {
    k_acvt<<<cdiv(Mr * Kd, 256), 256, 0, stream>>>(A, atmp, Mr * Kd);
    k_wconv<<<cdiv(Nc * Kd, 256), 256, 0, stream>>>(W, wtmp, Nc, Kd);
    dim3 grid(cdiv(Nc, 64), cdiv(Mr, 128));
    k_gemm_wmma<<<grid, 256, 0, stream>>>(atmp, wtmp, bias, C, Mr, Nc, Kd, relu);
  };

  // ---- prep ----
  k_flatten<<<cdiv(RE * Dm, 256), 256, 0, stream>>>(
      xin[0], xin[1], xin[2], xin[3], pin[0], pin[1], pin[2], pin[3],
      min_[0], min_[1], min_[2], min_[3], lvl_emb, xf, pf, mf);
  k_vr<<<1, 64, 0, stream>>>(min_[0], min_[1], min_[2], min_[3], vr);
  k_encref<<<cdiv(RE, 256), 256, 0, stream>>>(vr, refe);

  // ---- encoder (state in xf) ----
  for (int li = 0; li < NENCm; ++li) {
    const int P = ENC0 + li * 16;
    const float *val_w = F(P+0),  *val_b = F(P+1),  *off_w = F(P+2), *off_b = F(P+3);
    const float *aw_w  = F(P+4),  *aw_b  = F(P+5),  *out_w = F(P+6), *out_b = F(P+7);
    const float *n1g = F(P+8), *n1b = F(P+9), *n2g = F(P+10), *n2b = F(P+11);
    const float *l1w = F(P+12), *l1b = F(P+13), *l2w = F(P+14), *l2b = F(P+15);

    k_add<<<cdiv(RE * Dm, 256), 256, 0, stream>>>(xf, pf, qin, RE * Dm);
    gemm(xf,  val_w, val_b, valb, RE, Dm, Dm, 0);
    k_maskval<<<cdiv(RE * Dm, 256), 256, 0, stream>>>(valb, mf, RE);
    gemm(qin, off_w, off_b, offb, RE, Dm, Dm, 0);
    gemm(qin, aw_w,  aw_b,  awb,  RE, NHm * 16, Dm, 0);
    k_msda<<<RE, 256, 0, stream>>>(valb, offb, awb, refe, vr, samp, Ltok);
    gemm(samp, out_w, out_b, proj, RE, Dm, Dm, 0);
    k_addln<<<RE, Dm, 0, stream>>>(xf, proj, n1g, n1b, yb);
    gemm(yb,  l1w, l1b, ffh,  RE, DFFm, Dm, 1);
    gemm(ffh, l2w, l2b, proj, RE, Dm, DFFm, 0);
    k_addln<<<RE, Dm, 0, stream>>>(yb, proj, n2g, n2b, xf);
  }

  // ---- decoder prep ----
  k_split<<<cdiv(RQ * Dm, 256), 256, 0, stream>>>(qembed, qpos, xdec);
  k_refq<<<cdiv(RQ * 2, 256), 256, 0, stream>>>(qpos, ref_w, ref_b, refq);

  // ---- decoder layers ----
  for (int li = 0; li < NDECm; ++li) {
    const int P = DEC0 + li * 26;
    const float *q_w = F(P+0), *q_b = F(P+1), *k_w = F(P+2), *k_b = F(P+3);
    const float *v_w = F(P+4), *v_b = F(P+5), *sao_w = F(P+6), *sao_b = F(P+7);
    const float *mval_w = F(P+8),  *mval_b = F(P+9),  *moff_w = F(P+10), *moff_b = F(P+11);
    const float *maw_w  = F(P+12), *maw_b  = F(P+13), *mout_w = F(P+14), *mout_b = F(P+15);
    const float *n1g = F(P+16), *n1b = F(P+17), *n2g = F(P+18), *n2b = F(P+19);
    const float *n3g = F(P+20), *n3b = F(P+21);
    const float *l1w = F(P+22), *l1b = F(P+23), *l2w = F(P+24), *l2b = F(P+25);

    // self-attention
    k_add<<<cdiv(RQ * Dm, 256), 256, 0, stream>>>(xdec, qpos, qin, RQ * Dm);
    gemm(qin,  q_w, q_b, qb, RQ, Dm, Dm, 0);
    gemm(qin,  k_w, k_b, kb, RQ, Dm, Dm, 0);
    gemm(xdec, v_w, v_b, vb, RQ, Dm, Dm, 0);
    k_mha<<<Nb * NHm * NQm, 32, 0, stream>>>(qb, kb, vb, attb);
    gemm(attb, sao_w, sao_b, proj, RQ, Dm, Dm, 0);
    k_addln<<<RQ, Dm, 0, stream>>>(xdec, proj, n1g, n1b, x2q);

    // cross MSDA (value = encoder output xf)
    gemm(xf, mval_w, mval_b, valb, RE, Dm, Dm, 0);
    k_maskval<<<cdiv(RE * Dm, 256), 256, 0, stream>>>(valb, mf, RE);
    k_add<<<cdiv(RQ * Dm, 256), 256, 0, stream>>>(x2q, qpos, qin, RQ * Dm);
    gemm(qin, moff_w, moff_b, offb, RQ, Dm, Dm, 0);
    gemm(qin, maw_w,  maw_b,  awb,  RQ, NHm * 16, Dm, 0);
    k_msda<<<RQ, 256, 0, stream>>>(valb, offb, awb, refq, vr, samp, NQm);
    gemm(samp, mout_w, mout_b, proj, RQ, Dm, Dm, 0);
    k_addln<<<RQ, Dm, 0, stream>>>(x2q, proj, n2g, n2b, x3q);

    // FFN
    gemm(x3q, l1w, l1b, ffh,  RQ, DFFm, Dm, 1);
    gemm(ffh, l2w, l2b, proj, RQ, Dm, DFFm, 0);
    k_addln<<<RQ, Dm, 0, stream>>>(x3q, proj, n3g, n3b, xdec);

    // hs[li]
    k_copy<<<cdiv(RQ * Dm, 256), 256, 0, stream>>>(xdec, outp + (size_t)li * RQ * Dm,
                                                   RQ * Dm);
  }

  // init_ref + irefs
  k_refs_out<<<cdiv(RQ * 2, 256), 256, 0, stream>>>(refq, outp);
}
